// KernelAttention_83159156785689
// MI455X (gfx1250) — matrix operand order repaired and checked by reading, not compiled, not run on hardware
//
#include <hip/hip_runtime.h>

typedef __attribute__((ext_vector_type(16))) _Float16 v16h;
typedef __attribute__((ext_vector_type(8)))  _Float16 v8h;
typedef __attribute__((ext_vector_type(8)))  float    v8f;
typedef __attribute__((ext_vector_type(8)))  float    v8fv;

constexpr int   Bn = 2;
constexpr int   Tn = 2048;     // T == S
constexpr int   En = 1024;
constexpr int   Hn = 16;
constexpr int   Dn = 64;
constexpr int   Mn = Bn * Tn;  // 4096 rows for projection GEMMs
constexpr float QSCALE = 0.125f;  // D^-0.5

// ---------------------------------------------------------------------------
// WMMA helpers (CDNA5: v_wmma_f32_16x16x32_f16, D = A(16x32) * B(32x16) + C)
// A layout: lane L (row M=L%16) holds K = 16*h..16*h+15 where h = L/16
// -> one 32B chunk per lane (same K order as B).
// B layout: lane L holds column N=L%16, K rows 16*(L/16)..+15 -> one 32B chunk.
// C/D: VGPR r, lane L -> row M = r + 8*(L/16), col N = L%16.
// ---------------------------------------------------------------------------
__device__ __forceinline__ v8f wmma_f16(v16h a, v16h b, v8f c) {
  return __builtin_amdgcn_wmma_f32_16x16x32_f16(false, a, false, b, (short)0, c,
                                                false, false);
}

__device__ __forceinline__ v16h load_a(const _Float16* row, int half) {
  v16h a = *(const v16h*)(row + half * 16);
  return a;
}

// ---------------------------------------------------------------------------
// Bulk f32 -> f16 conversion (amortizes all cvt work out of GEMM hot loops).
// One v8 load / v8 store per thread; n must be a multiple of 8*256.
// ---------------------------------------------------------------------------
__global__ void __launch_bounds__(256)
cvt_f16(const float* __restrict__ in, _Float16* __restrict__ out) {
  const size_t i = ((size_t)blockIdx.x * 256 + threadIdx.x) * 8;
  v8fv x = *(const v8fv*)(in + i);
  v8h y;
#pragma unroll
  for (int j = 0; j < 8; ++j) y[j] = (_Float16)x[j];
  *(v8h*)(out + i) = y;
}

// ---------------------------------------------------------------------------
// Projection GEMM (all-f16 operands): out[m,n] = (X @ W^T + bias) * scale
// Each wave computes a 32(M) x 64(N) strip: two A tiles share every B tile
// (8 WMMAs per 4 B-tile loads). A 64-wide N strip == one head.
// MODE 0: f16 row-major out + per-(b,h,t) sum-of-squares (for q / k)
// MODE 1: f16 transposed out [b,h,d,s] (for v -> PV B-matrix layout)
// MODE 2: f32 row-major out (final Wo projection)
// ---------------------------------------------------------------------------
template <int MODE>
__global__ void __launch_bounds__(256)
proj_gemm(const _Float16* __restrict__ X, const _Float16* __restrict__ W,
          const float* __restrict__ bias, _Float16* __restrict__ outh,
          float* __restrict__ outf, float* __restrict__ sq, float scale) {
  const int lane = threadIdx.x & 31;
  const int half = lane >> 4;
  const int lm   = lane & 15;
  const int wid  = blockIdx.x * 8 + (threadIdx.x >> 5);
  const int mt   = wid & (Mn / 32 - 1);   // 128 m-tiles (32 rows each)
  const int ns   = wid >> 7;              // 16 n-strips (== head id)
  const int m0   = mt * 32;
  const int n0   = ns * 64;

  v8f acc[2][4] = {};
  const _Float16* arow0 = X + (size_t)(m0 + lm) * En;
  const _Float16* arow1 = arow0 + (size_t)16 * En;

  for (int k0 = 0; k0 < En; k0 += 32) {
    v16h a0 = load_a(arow0 + k0, half);
    v16h a1 = load_a(arow1 + k0, half);
#pragma unroll
    for (int sub = 0; sub < 4; ++sub) {
      // B[k,n] = W[n,k]; lane's column is a contiguous 32B row chunk of W.
      v16h bm =
          *(const v16h*)(W + (size_t)(n0 + sub * 16 + lm) * En + k0 + half * 16);
      acc[0][sub] = wmma_f16(a0, bm, acc[0][sub]);
      acc[1][sub] = wmma_f16(a1, bm, acc[1][sub]);
    }
  }

  float sqp[2][8];
#pragma unroll
  for (int g = 0; g < 2; ++g)
#pragma unroll
    for (int r = 0; r < 8; ++r) sqp[g][r] = 0.0f;

#pragma unroll
  for (int g = 0; g < 2; ++g) {
#pragma unroll
    for (int sub = 0; sub < 4; ++sub) {
      const int n = n0 + sub * 16 + lm;
      const float bn = bias[n];
      if constexpr (MODE == 1) {
        // Transposed [b,h,d,s]: lane's 8 rows are consecutive s -> one v8h.
        v8h pk;
#pragma unroll
        for (int r = 0; r < 8; ++r)
          pk[r] = (_Float16)((acc[g][sub][r] + bn) * scale);
        const int bb = m0 / Tn;
        const int s0 = (m0 % Tn) + g * 16 + 8 * half;
        size_t idx =
            (((size_t)bb * Hn + ns) * Dn + (sub * 16 + lm)) * (size_t)Tn + s0;
        *(v8h*)(outh + idx) = pk;
      } else {
#pragma unroll
        for (int r = 0; r < 8; ++r) {
          const float c = (acc[g][sub][r] + bn) * scale;
          const size_t row = (size_t)(m0 + g * 16 + r + 8 * half);
          if constexpr (MODE == 0) {
            outh[row * En + n] = (_Float16)c;
            sqp[g][r] += c * c;
          } else {  // MODE == 2
            outf[row * En + n] = c;
          }
        }
      }
    }
  }

  if constexpr (MODE == 0) {
    // Reduce ||row||^2 across the 16 lanes of each half-wave (64 cols total).
#pragma unroll
    for (int m = 1; m < 16; m <<= 1)
#pragma unroll
      for (int g = 0; g < 2; ++g)
#pragma unroll
        for (int r = 0; r < 8; ++r)
          sqp[g][r] += __shfl_xor(sqp[g][r], m, 32);
    if (lm == 0) {
      const int bb = m0 / Tn;
#pragma unroll
      for (int g = 0; g < 2; ++g) {
        const int t0 = (m0 % Tn) + g * 16 + 8 * half;
#pragma unroll
        for (int r = 0; r < 8; ++r)
          sq[((size_t)bb * Hn + ns) * Tn + t0 + r] = sqp[g][r];
      }
    }
  }
}

// ---------------------------------------------------------------------------
// Gaussian-kernel attention per (b, h, 64-row t-tile). 4 waves, 16 rows each.
// Pass 1: row sums of exp(-0.5*(||q||^2+||k||^2-2 q.k)) via WMMA QK^T.
// Pass 2: recompute, store normalized probs to d_out (the only 537 MB write),
//         bounce probs through LDS into WMMA A layout, accumulate PV.
// ---------------------------------------------------------------------------
__global__ void __launch_bounds__(128)
gauss_attn(const _Float16* __restrict__ qp, const _Float16* __restrict__ kp,
           const _Float16* __restrict__ vpT, const float* __restrict__ q2g,
           const float* __restrict__ k2g, float* __restrict__ attn_out,
           _Float16* __restrict__ attnh) {
  const int lane = threadIdx.x & 31;
  const int half = lane >> 4;
  const int lm   = lane & 15;
  const int wave = threadIdx.x >> 5;
  const int tt   = blockIdx.x & (Tn / 64 - 1);
  const int h    = (blockIdx.x >> 5) & (Hn - 1);
  const int b    = blockIdx.x >> 9;
  const int t0   = tt * 64 + wave * 16;

  __shared__ __align__(16) _Float16 smem[4][16 * 32];
  _Float16* ldsw = smem[wave];

  // Q tile (16 x 64) held in registers for both passes.
  const _Float16* qrow = qp + (size_t)(b * Tn + t0 + lm) * En + h * Dn;
  const v16h aq0 = load_a(qrow, half);
  const v16h aq1 = load_a(qrow + 32, half);

  const float* q2b = q2g + ((size_t)b * Hn + h) * Tn + t0 + 8 * half;
  float q2r[8];
#pragma unroll
  for (int r = 0; r < 8; ++r) q2r[r] = q2b[r];

  const float*    k2b   = k2g + ((size_t)b * Hn + h) * Tn;
  const _Float16* kbase = kp + (size_t)b * Tn * En + h * Dn;

  // ---- pass 1: row sums -------------------------------------------------
  float rs[8] = {};
  for (int s0 = 0; s0 < Tn; s0 += 16) {
    const _Float16* krow = kbase + (size_t)(s0 + lm) * En + half * 16;
    v16h bk0 = *(const v16h*)(krow);
    v16h bk1 = *(const v16h*)(krow + 32);
    v8f qk = {};
    qk = wmma_f16(aq0, bk0, qk);
    qk = wmma_f16(aq1, bk1, qk);
    const float k2c = k2b[s0 + lm];
#pragma unroll
    for (int r = 0; r < 8; ++r) {
      float d2 = q2r[r] + k2c - 2.0f * qk[r];
      rs[r] += __expf(-0.5f * d2);
    }
    if (s0 + 16 < Tn) __builtin_prefetch(krow + (size_t)16 * En, 0, 1);
  }
#pragma unroll
  for (int m = 1; m < 16; m <<= 1)
#pragma unroll
    for (int r = 0; r < 8; ++r) rs[r] += __shfl_xor(rs[r], m, 32);
  float inv[8];
#pragma unroll
  for (int r = 0; r < 8; ++r) inv[r] = 1.0f / rs[r];

  // ---- pass 2: normalized probs out + PV accumulation -------------------
  v8f acc[4] = {};
  float* aout = attn_out + (((size_t)b * Hn + h) * Tn + t0) * (size_t)Tn;
  const _Float16* vbase = vpT + (((size_t)b * Hn + h) * Dn) * (size_t)Tn;

  for (int s0 = 0; s0 < Tn; s0 += 32) {
#pragma unroll
    for (int j = 0; j < 2; ++j) {
      const int s = s0 + j * 16;
      const _Float16* krow = kbase + (size_t)(s + lm) * En + half * 16;
      v16h bk0 = *(const v16h*)(krow);
      v16h bk1 = *(const v16h*)(krow + 32);
      v8f qk = {};
      qk = wmma_f16(aq0, bk0, qk);
      qk = wmma_f16(aq1, bk1, qk);
      const float k2c = k2b[s + lm];
#pragma unroll
      for (int r = 0; r < 8; ++r) {
        float d2 = q2r[r] + k2c - 2.0f * qk[r];
        float wn = __expf(-0.5f * d2) * inv[r];
        const int row = r + 8 * half;
        aout[(size_t)row * Tn + s + lm] = wn;              // kernel_attn (f32)
        ldsw[row * 32 + j * 16 + lm] = (_Float16)wn;       // stage for PV
      }
    }
    // Wave-local LDS RAW fence: C-layout stores -> A-layout loads.
    asm volatile("s_wait_dscnt 0x0" ::: "memory");
    v16h ap = load_a(ldsw + lm * 32, half);
#pragma unroll
    for (int d = 0; d < 4; ++d) {
      const _Float16* vrow = vbase + (size_t)(d * 16 + lm) * Tn + s0 + half * 16;
      v16h bv = *(const v16h*)vrow;
      acc[d] = wmma_f16(ap, bv, acc[d]);
    }
  }

  // attn result back to [b, t, (h d)] in f16 for the Wo GEMM.
#pragma unroll
  for (int d = 0; d < 4; ++d)
#pragma unroll
    for (int r = 0; r < 8; ++r)
      attnh[(size_t)(b * Tn + t0 + r + 8 * half) * En + h * Dn + d * 16 + lm] =
          (_Float16)acc[d][r];
}

// ---------------------------------------------------------------------------
extern "C" void kernel_launch(void* const* d_in, const int* in_sizes, int n_in,
                              void* d_out, int out_size, void* d_ws,
                              size_t ws_size, hipStream_t stream) {
  (void)in_sizes; (void)n_in; (void)out_size; (void)ws_size;
  const float* q  = (const float*)d_in[0];
  const float* k  = (const float*)d_in[1];
  const float* v  = (const float*)d_in[2];
  const float* Wq = (const float*)d_in[3];
  const float* bq = (const float*)d_in[4];
  const float* Wk = (const float*)d_in[5];
  const float* bk = (const float*)d_in[6];
  const float* Wv = (const float*)d_in[7];
  const float* bv = (const float*)d_in[8];
  const float* Wo = (const float*)d_in[9];
  const float* bo = (const float*)d_in[10];

  float* out_f = (float*)d_out;                      // (B,T,E) f32
  float* kattn = out_f + (size_t)Bn * Tn * En;       // (B,H,T,S) f32

  char* ws = (char*)d_ws;
  _Float16* qph   = (_Float16*)(ws);                 //  8 MB
  _Float16* kph   = (_Float16*)(ws + (8u << 20));    //  8 MB
  _Float16* vpT   = (_Float16*)(ws + (16u << 20));   //  8 MB [b,h,d,s]
  _Float16* attnh = (_Float16*)(ws + (24u << 20));   //  8 MB
  float*    q2    = (float*)(ws + (32u << 20));      // 256 KB
  float*    k2    = (float*)(ws + (33u << 20));      // 256 KB
  _Float16* q16   = (_Float16*)(ws + (34u << 20));   //  8 MB
  _Float16* k16   = (_Float16*)(ws + (42u << 20));   //  8 MB
  _Float16* v16s  = (_Float16*)(ws + (50u << 20));   //  8 MB
  _Float16* Wq16  = (_Float16*)(ws + (58u << 20));   //  2 MB
  _Float16* Wk16  = (_Float16*)(ws + (60u << 20));   //  2 MB
  _Float16* Wv16  = (_Float16*)(ws + (62u << 20));   //  2 MB
  _Float16* Wo16  = (_Float16*)(ws + (64u << 20));   //  2 MB

  // One-shot f32 -> f16 conversions (activations + weights).
  const int XB = (int)((size_t)Mn * En / (8 * 256));  // 2048 blocks
  const int WB = (int)((size_t)En * En / (8 * 256));  // 512 blocks
  cvt_f16<<<XB, 256, 0, stream>>>(q, q16);
  cvt_f16<<<XB, 256, 0, stream>>>(k, k16);
  cvt_f16<<<XB, 256, 0, stream>>>(v, v16s);
  cvt_f16<<<WB, 256, 0, stream>>>(Wq, Wq16);
  cvt_f16<<<WB, 256, 0, stream>>>(Wk, Wk16);
  cvt_f16<<<WB, 256, 0, stream>>>(Wv, Wv16);
  cvt_f16<<<WB, 256, 0, stream>>>(Wo, Wo16);

  const dim3 pg(256), pb(256);  // 2048 waves: 128 m-tiles x 16 n-strips
  proj_gemm<0><<<pg, pb, 0, stream>>>(q16, Wq16, bq, qph, nullptr, q2, QSCALE);
  proj_gemm<0><<<pg, pb, 0, stream>>>(k16, Wk16, bk, kph, nullptr, k2, 1.0f);
  proj_gemm<1><<<pg, pb, 0, stream>>>(v16s, Wv16, bv, vpT, nullptr, nullptr,
                                      1.0f);
  gauss_attn<<<dim3(Bn * Hn * (Tn / 64)), dim3(128), 0, stream>>>(
      qph, kph, vpT, q2, k2, kattn, attnh);
  proj_gemm<2><<<pg, pb, 0, stream>>>(attnh, Wo16, bo, nullptr, out_f, nullptr,
                                      1.0f);
}
